// Mamba3Simple_81406810128671
// MI455X (gfx1250) — compile-verified, matrix operand order checked by reading, not run on hardware
//
#include <hip/hip_runtime.h>

typedef __bf16 bf16_t;
typedef __attribute__((ext_vector_type(16))) __bf16 v16bf;
typedef __attribute__((ext_vector_type(8)))  __bf16 v8bf;
typedef __attribute__((ext_vector_type(8)))  float  v8f;
typedef int v4i_ __attribute__((vector_size(16)));   // matches builtin param type

#define AS1 __attribute__((address_space(1)))
#define AS3 __attribute__((address_space(3)))

#define H_      32
#define P_      64
#define N_      64
#define DM_     1024
#define L_      2048
#define BSZ_    2
#define DI_     2048              // H*P
#define TH_     1024              // H*N/2
#define DPROJ_  5312
#define PS_     5376              // proj row stride, padded to 128-col GEMM tiles
#define ROWS_   (BSZ_*L_)         // 4096

// column offsets inside one proj row
#define OFF_Z     0
#define OFF_X     DI_
#define OFF_B     (2*DI_)
#define OFF_C     (2*DI_ + N_)
#define OFF_DT    (2*DI_ + 2*N_)
#define OFF_THETA (OFF_DT + H_)
#define OFF_LAM   (OFF_THETA + TH_)

// LDS staging geometry: rows padded 64B -> 80B so 32-lane x 16B reads are
// conflict-free (each bank hit exactly twice = minimum for 512B/instr)
#define LDS_ROW_   80
#define ATILE_BY_  (64 * LDS_ROW_)    //  5120 B
#define BTILE_BY_  (128 * LDS_ROW_)   // 10240 B

// ---------------------------------------------------------------- converts
__global__ __launch_bounds__(256) void k_cvt_bf16(const float* __restrict__ s,
                                                  bf16_t* __restrict__ d, int n) {
  int i = blockIdx.x * blockDim.x + threadIdx.x;
  if (i < n) d[i] = (bf16_t)s[i];
}

// in_proj_w (5312 x 1024) -> bf16 padded to 5376 rows (pad rows = 0)
__global__ __launch_bounds__(256) void k_cvt_w1(const float* __restrict__ s,
                                                bf16_t* __restrict__ d) {
  int i = blockIdx.x * blockDim.x + threadIdx.x;   // over PS_*DM_
  int row = i >> 10;                                // /1024
  int col = i & 1023;
  float v = (row < DPROJ_) ? s[row * DM_ + col] : 0.0f;
  d[i] = (bf16_t)v;
}

// ---------------------------------------------------------------- async copy
// 16B global -> LDS. Async path uses the CDNA5 GLOBAL_LOAD_ASYNC_TO_LDS_B128
// (ASYNCcnt-tracked DMA); fallback stages through VGPRs + ds_store.
__device__ __forceinline__ void copy16_g2l(void* lds, const void* g) {
#if __has_builtin(__builtin_amdgcn_global_load_async_to_lds_b128)
  __builtin_amdgcn_global_load_async_to_lds_b128(
      (AS1 v4i_*)(unsigned long long)g,
      (AS3 v4i_*)(unsigned int)(unsigned long long)lds,
      0, 0);
#else
  *(v8bf*)lds = *(const v8bf*)g;
#endif
}

__device__ __forceinline__ void async_wait_barrier() {
#if __has_builtin(__builtin_amdgcn_s_wait_asynccnt)
  __builtin_amdgcn_s_wait_asynccnt(0);
#endif
  __syncthreads();
}

// Stage one K-step: A-tile 64x32 bf16 (64B/row, 1 chunk/thread),
// B-tile 128x32 bf16 (64B/row, 2 chunks/thread). 256 threads.
__device__ __forceinline__ void stage_tiles(const bf16_t* __restrict__ A,
                                            const bf16_t* __restrict__ Bm,
                                            int rowBase, int colBase, int k0,
                                            int K, char* sA, char* sB, int tid) {
  int ar = tid >> 2, ac = tid & 3;
  copy16_g2l(sA + ar * LDS_ROW_ + ac * 16,
             (const char*)(A + (size_t)(rowBase + ar) * K + k0) + ac * 16);
#pragma unroll
  for (int j = 0; j < 2; ++j) {
    int idx = tid + j * 256;
    int br = idx >> 2, bc = idx & 3;
    copy16_g2l(sB + br * LDS_ROW_ + bc * 16,
               (const char*)(Bm + (size_t)(colBase + br) * K + k0) + bc * 16);
  }
}

// ---------------------------------------------------------------- WMMA GEMM
// C[M,N] = A[M,K] * B[N,K]^T, tiles exact (M%64==0, N%128==0, K%32==0).
// 16-bit A fragment layout (ISA 7.12.2): lane m (0..15) holds K = half*8..+7
// in v0..3 and K = 16+half*8..+7 in v4..7, half = lane>=16. B uses the
// mirrored layout with lane indexing N.
__device__ __forceinline__ v16bf lds_frag(const char* base, int row, int half) {
  const char* p = base + row * LDS_ROW_ + half * 16;
  v8bf lo = *(const v8bf*)p;          // K chunk at half*8
  v8bf hi = *(const v8bf*)(p + 32);   // K chunk at 16 + half*8
  return __builtin_shufflevector(lo, hi, 0,1,2,3,4,5,6,7,8,9,10,11,12,13,14,15);
}

__global__ __launch_bounds__(256) void k_gemm_bf16(
    const bf16_t* __restrict__ A, const bf16_t* __restrict__ Bm,
    float* __restrict__ C, int M, int N, int K, int ldc) {
  __shared__ __align__(16) char smem[2 * ATILE_BY_ + 2 * BTILE_BY_]; // 30 KB
  char* sA0 = smem;
  char* sA1 = smem + ATILE_BY_;
  char* sB0 = smem + 2 * ATILE_BY_;
  char* sB1 = smem + 2 * ATILE_BY_ + BTILE_BY_;

  int tid  = threadIdx.x;
  int lane = tid & 31, wid = tid >> 5;
  int wm = wid >> 2, wn = wid & 3;            // 2 x 4 wave grid
  int rowBase = blockIdx.x * 64;
  int colBase = blockIdx.y * 128;
  int lr = lane & 15, lh = lane >> 4;

  stage_tiles(A, Bm, rowBase, colBase, 0, K, sA0, sB0, tid);
  async_wait_barrier();

  v8f acc00 = {}, acc01 = {}, acc10 = {}, acc11 = {};
  int nsteps = K >> 5;
  for (int s = 0; s < nsteps; ++s) {
    char* cA = (s & 1) ? sA1 : sA0;
    char* cB = (s & 1) ? sB1 : sB0;
    char* nA = (s & 1) ? sA0 : sA1;
    char* nB = (s & 1) ? sB0 : sB1;
    if (s + 1 < nsteps)   // prefetch next K-step while computing this one
      stage_tiles(A, Bm, rowBase, colBase, (s + 1) << 5, K, nA, nB, tid);

    v16bf a0 = lds_frag(cA, wm * 32      + lr, lh);
    v16bf a1 = lds_frag(cA, wm * 32 + 16 + lr, lh);
    v16bf b0 = lds_frag(cB, wn * 32      + lr, lh);
    v16bf b1 = lds_frag(cB, wn * 32 + 16 + lr, lh);
    acc00 = __builtin_amdgcn_wmma_f32_16x16x32_bf16(false, a0, false, b0, (short)0, acc00, false, false);
    acc01 = __builtin_amdgcn_wmma_f32_16x16x32_bf16(false, a0, false, b1, (short)0, acc01, false, false);
    acc10 = __builtin_amdgcn_wmma_f32_16x16x32_bf16(false, a1, false, b0, (short)0, acc10, false, false);
    acc11 = __builtin_amdgcn_wmma_f32_16x16x32_bf16(false, a1, false, b1, (short)0, acc11, false, false);

    async_wait_barrier();
  }

  // C/D layout: VGPR r, lane l -> row = (l>>4)*8 + r, col = l&15
  int r0 = rowBase + wm * 32 + lh * 8;
  int c0 = colBase + wn * 32 + lr;
  #pragma unroll
  for (int r = 0; r < 8; ++r) {
    C[(size_t)(r0 + r)      * ldc + c0]      = acc00[r];
    C[(size_t)(r0 + r)      * ldc + c0 + 16] = acc01[r];
    C[(size_t)(r0 + 16 + r) * ldc + c0]      = acc10[r];
    C[(size_t)(r0 + 16 + r) * ldc + c0 + 16] = acc11[r];
  }
}

// ---------------------------------------------------------------- RMS scales
__global__ __launch_bounds__(32) void k_rmsinv(const float* __restrict__ proj,
                                               float* __restrict__ invB,
                                               float* __restrict__ invC) {
  int row = blockIdx.x;
  int lane = threadIdx.x;
  const float* pr = proj + (size_t)row * PS_;
  float b0 = pr[OFF_B + lane], b1 = pr[OFF_B + 32 + lane];
  float c0 = pr[OFF_C + lane], c1 = pr[OFF_C + 32 + lane];
  float sb = b0 * b0 + b1 * b1;
  float sc = c0 * c0 + c1 * c1;
  for (int o = 16; o > 0; o >>= 1) {
    sb += __shfl_xor(sb, o, 32);
    sc += __shfl_xor(sc, o, 32);
  }
  if (lane == 0) {
    invB[row] = rsqrtf(sb * (1.0f / N_) + 1e-5f);
    invC[row] = rsqrtf(sc * (1.0f / N_) + 1e-5f);
  }
}

// ---------------------------------------------------------------- scan
// One block per (b,h). State h[P][N] + prevBx in registers: 16 f32 each per
// thread (thread t: p = t>>2, n = (t&3)*16 .. +15). Fuses theta cumsum, RoPE,
// B/C scale+bias, dt/alpha/gamma/beta and the recurrence + D residual.
__global__ __launch_bounds__(256) void k_scan(
    const float* __restrict__ proj, const float* __restrict__ invB,
    const float* __restrict__ invC, const float* __restrict__ dt_bias,
    const float* __restrict__ A_log, const float* __restrict__ Dp,
    const float* __restrict__ Bnw, const float* __restrict__ Cnw,
    const float* __restrict__ Bbias, const float* __restrict__ Cbias,
    float* __restrict__ y_raw) {
  int b = blockIdx.x >> 5;
  int h = blockIdx.x & 31;
  int tid = threadIdx.x;
  int p  = tid >> 2;
  int nb = (tid & 3) * 16;

  __shared__ float sx[64], sBp[64], sCp[64], sB[64], sC[64], sc_[32], ss_[32], sScal[3];

  float hreg[16], pBx[16];
  #pragma unroll
  for (int i = 0; i < 16; ++i) { hreg[i] = 0.0f; pBx[i] = 0.0f; }

  float tc  = 0.0f;                 // theta cumsum (only threads 192..223 use)
  float Dh  = Dp[h];
  float dtb = dt_bias[h];
  float Ah  = -expf(A_log[h]);

  for (int l = 0; l < L_; ++l) {
    int row = b * L_ + l;
    const float* pr = proj + (size_t)row * PS_;
    if (tid < 64) {
      sx[tid] = pr[OFF_X + h * P_ + tid];
    } else if (tid < 128) {
      int n = tid - 64;
      sBp[n] = pr[OFF_B + n] * invB[row] * Bnw[n] + Bbias[h * N_ + n];
    } else if (tid < 192) {
      int n = tid - 128;
      sCp[n] = pr[OFF_C + n] * invC[row] * Cnw[n] + Cbias[h * N_ + n];
    } else if (tid < 224) {
      int j = tid - 192;
      tc += pr[OFF_THETA + h * (N_ / 2) + j];
      sc_[j] = cosf(tc);
      ss_[j] = sinf(tc);
    } else if (tid == 224) {
      float dtr = pr[OFF_DT + h] + dtb;
      float dt  = (dtr > 20.0f) ? dtr : log1pf(expf(dtr));     // softplus
      float alpha = expf(dt * Ah);
      float lam   = 1.0f / (1.0f + expf(-pr[OFF_LAM + h]));
      sScal[0] = alpha;
      sScal[1] = lam * dt;
      sScal[2] = (1.0f - lam) * dt * alpha;
    }
    __syncthreads();
    if (tid < 32) {                  // RoPE on B
      int j = tid;
      float v1 = sBp[j], v2 = sBp[j + 32], cj = sc_[j], sj = ss_[j];
      sB[j]      = v1 * cj - v2 * sj;
      sB[j + 32] = v1 * sj + v2 * cj;
    } else if (tid < 64) {           // RoPE on C
      int j = tid - 32;
      float v1 = sCp[j], v2 = sCp[j + 32], cj = sc_[j], sj = ss_[j];
      sC[j]      = v1 * cj - v2 * sj;
      sC[j + 32] = v1 * sj + v2 * cj;
    }
    __syncthreads();
    float alpha = sScal[0], gamma = sScal[1], beta = sScal[2];
    float x = sx[p];
    float y = 0.0f;
    #pragma unroll
    for (int i = 0; i < 16; ++i) {
      int n = nb + i;
      float Bx = x * sB[n];
      float hn = alpha * hreg[i] + gamma * Bx + beta * pBx[i];
      hreg[i] = hn;
      pBx[i]  = Bx;
      y += hn * sC[n];
    }
    y += __shfl_xor(y, 1, 32);       // reduce over the 4 lanes sharing p
    y += __shfl_xor(y, 2, 32);
    if ((tid & 3) == 0) y_raw[(size_t)row * DI_ + h * P_ + p] = y + x * Dh;
    __syncthreads();
  }
}

// ---------------------------------------------------------------- gate + RMS + bf16
__global__ __launch_bounds__(256) void k_gate(
    const float* __restrict__ proj, const float* __restrict__ y_raw,
    const float* __restrict__ norm_w, bf16_t* __restrict__ y_bf) {
  int row = blockIdx.x;
  int tid = threadIdx.x;
  const float* pz = proj  + (size_t)row * PS_ + OFF_Z;
  const float* py = y_raw + (size_t)row * DI_;
  float vals[8];
  float ss = 0.0f;
  #pragma unroll
  for (int j = 0; j < 8; ++j) {
    int i = tid + j * 256;
    float z = pz[i];
    float v = py[i] * (z / (1.0f + expf(-z)));   // y * silu(z)
    vals[j] = v;
    ss += v * v;
  }
  for (int o = 16; o > 0; o >>= 1) ss += __shfl_xor(ss, o, 32);
  __shared__ float red[8];
  int lane = tid & 31, wid = tid >> 5;
  if (lane == 0) red[wid] = ss;
  __syncthreads();
  if (tid == 0) {
    float t = 0.0f;
    for (int w = 0; w < 8; ++w) t += red[w];
    red[0] = rsqrtf(t * (1.0f / DI_) + 1e-5f);
  }
  __syncthreads();
  float scale = red[0];
  #pragma unroll
  for (int j = 0; j < 8; ++j) {
    int i = tid + j * 256;
    y_bf[(size_t)row * DI_ + i] = (bf16_t)(vals[j] * scale * norm_w[i]);
  }
}

// ---------------------------------------------------------------- launch
extern "C" void kernel_launch(void* const* d_in, const int* in_sizes, int n_in,
                              void* d_out, int out_size, void* d_ws, size_t ws_size,
                              hipStream_t stream) {
  const float* u       = (const float*)d_in[0];
  const float* w1      = (const float*)d_in[1];
  const float* dt_bias = (const float*)d_in[2];
  const float* A_log   = (const float*)d_in[3];
  const float* Dp      = (const float*)d_in[4];
  const float* Bnw     = (const float*)d_in[5];
  const float* Cnw     = (const float*)d_in[6];
  const float* Bbias   = (const float*)d_in[7];
  const float* Cbias   = (const float*)d_in[8];
  const float* norm_w  = (const float*)d_in[9];
  const float* w2      = (const float*)d_in[10];
  float* out = (float*)d_out;
  (void)in_sizes; (void)n_in; (void)out_size; (void)ws_size;

  char* ws = (char*)d_ws;
  size_t o = 0;
  bf16_t* u_bf  = (bf16_t*)(ws + o); o += (size_t)ROWS_ * DM_ * 2;   //  8 MB
  bf16_t* w1_bf = (bf16_t*)(ws + o); o += (size_t)PS_   * DM_ * 2;   // 11 MB
  bf16_t* w2_bf = (bf16_t*)(ws + o); o += (size_t)DM_   * DI_ * 2;   //  4 MB
  float*  proj  = (float*) (ws + o); o += (size_t)ROWS_ * PS_ * 4;   // 88 MB
  float*  invB  = (float*) (ws + o); o += (size_t)ROWS_ * 4;
  float*  invC  = (float*) (ws + o); o += (size_t)ROWS_ * 4;
  float*  y_raw = (float*) (ws + o); o += (size_t)ROWS_ * DI_ * 4;   // 33 MB
  bf16_t* y_bf  = (bf16_t*)(ws + o); o += (size_t)ROWS_ * DI_ * 2;   // 17 MB

  // precision converts (pad in_proj_w rows 5312 -> 5376 with zeros)
  k_cvt_bf16<<<(ROWS_ * DM_) / 256, 256, 0, stream>>>(u, u_bf, ROWS_ * DM_);
  k_cvt_w1  <<<(PS_ * DM_)   / 256, 256, 0, stream>>>(w1, w1_bf);
  k_cvt_bf16<<<(DM_ * DI_)   / 256, 256, 0, stream>>>(w2, w2_bf, DM_ * DI_);

  // GEMM1: proj[4096 x 5376] = u[4096 x 1024] * W1^T
  dim3 g1(ROWS_ / 64, PS_ / 128);
  k_gemm_bf16<<<g1, 256, 0, stream>>>(u_bf, w1_bf, proj, ROWS_, PS_, DM_, PS_);

  // per-row RMS scales for Braw / Craw
  k_rmsinv<<<ROWS_, 32, 0, stream>>>(proj, invB, invC);

  // sequential SSM scan, one block per (b,h)
  k_scan<<<BSZ_ * H_, 256, 0, stream>>>(proj, invB, invC, dt_bias, A_log, Dp,
                                        Bnw, Cnw, Bbias, Cbias, y_raw);

  // gating + RMS norm + bf16 convert
  k_gate<<<ROWS_, 256, 0, stream>>>(proj, y_raw, norm_w, y_bf);

  // GEMM2: out[4096 x 1024] = y[4096 x 2048] * W2^T
  dim3 g2(ROWS_ / 64, DM_ / 128);
  k_gemm_bf16<<<g2, 256, 0, stream>>>(y_bf, w2_bf, out, ROWS_, DM_, DI_, DM_);
}
